// RotaryPEMultiHeadAttention_29454885715995
// MI455X (gfx1250) — matrix operand order repaired and checked by reading, not compiled, not run on hardware
//
#include <hip/hip_runtime.h>
#include <hip/hip_bf16.h>
#include <math.h>

// ---------------------------------------------------------------------------
// RoPE multi-head attention for MI455X (gfx1250, wave32, WMMA bf16)
//
// ~137 GFLOP total; ~200MB working set fits global L2 (192MB); HBM time ~5us
// @ 23.3 TB/s => matrix-compute bound. All matmuls on v_wmma_f32_16x16x32_bf16
// with fp32 accumulation; softmax fp32 VALU with DPP row reductions; weight
// slabs staged to LDS with gfx1250 async loads (ASYNCcnt).
// ---------------------------------------------------------------------------

#define HID   1024
#define HEADS 16
#define KDIM  64
#define SEQ   2048
#define BATCH 4
#define NTOK  (BATCH * SEQ)       // 8192 tokens
#define SCALE 0.125f              // 1/sqrt(64)
// ln(10000)/16  (theta_i = 10000^(-i/16))
#define LN_BASE_OVER_16 0.57564627324851149f

typedef __attribute__((ext_vector_type(16))) __bf16 v16bf;
typedef __attribute__((ext_vector_type(8)))  __bf16 v8bf;
typedef __attribute__((ext_vector_type(8)))  float  v8f;
typedef __attribute__((ext_vector_type(4)))  int    v4i;

#if __has_builtin(__builtin_amdgcn_global_load_async_to_lds_b128) && \
    __has_builtin(__builtin_amdgcn_s_wait_asynccnt)
#define USE_ASYNC_LDS 1
#else
#define USE_ASYNC_LDS 0
#endif

static __device__ __forceinline__ v8f wmma_bf16(v16bf a, v16bf b, v8f c) {
  // (neg_a, A, neg_b, B, c_mod, C, reuse_a, reuse_b)
  return __builtin_amdgcn_wmma_f32_16x16x32_bf16(false, a, false, b, (short)0,
                                                 c, false, false);
}

static __device__ __forceinline__ v16bf cat8(v8bf r0, v8bf r1) {
  v16bf a;
#pragma unroll
  for (int e = 0; e < 8; ++e) { a[e] = r0[e]; a[8 + e] = r1[e]; }
  return a;
}

// 16B global -> LDS, async (gfx1250 GLOBAL_LOAD_ASYNC_TO_LDS_B128) when
// available; otherwise plain load + ds_store (ordered by __syncthreads).
static __device__ __forceinline__ void async_copy16(const __bf16* g,
                                                    __bf16* l) {
#if USE_ASYNC_LDS
  void* gnc = const_cast<__bf16*>(g);
  __builtin_amdgcn_global_load_async_to_lds_b128(
      (__attribute__((address_space(1))) v4i*)gnc,
      (__attribute__((address_space(3))) v4i*)l, 0, 0);
#else
  *(v8bf*)l = *(const v8bf*)g;
#endif
}
static __device__ __forceinline__ void async_wait0() {
#if USE_ASYNC_LDS
  __builtin_amdgcn_s_wait_asynccnt(0);
#endif
}
static __device__ __forceinline__ void async_wait1() {
#if USE_ASYNC_LDS
  __builtin_amdgcn_s_wait_asynccnt(1);
#endif
}

// max-butterfly across a 16-lane half using DPP (no LDS traffic)
template <int CTRL>
static __device__ __forceinline__ float max_dpp(float x) {
  const int y = __builtin_amdgcn_update_dpp(0, __builtin_bit_cast(int, x),
                                            CTRL, 0xF, 0xF, true);
  return fmaxf(x, __builtin_bit_cast(float, y));
}
static __device__ __forceinline__ float rowmax16(float x) {
#if __has_builtin(__builtin_amdgcn_update_dpp)
  x = max_dpp<0xB1>(x);    // quad_perm [1,0,3,2]  (xor 1)
  x = max_dpp<0x4E>(x);    // quad_perm [2,3,0,1]  (xor 2)
  x = max_dpp<0x141>(x);   // row_half_mirror      (xor across quads in 8)
  x = max_dpp<0x140>(x);   // row_mirror           (xor across 8s in 16)
#else
  x = fmaxf(x, __shfl_xor(x, 1, 32));
  x = fmaxf(x, __shfl_xor(x, 2, 32));
  x = fmaxf(x, __shfl_xor(x, 4, 32));
  x = fmaxf(x, __shfl_xor(x, 8, 32));
#endif
  return x;
}

// ---------------------------------------------------------------------------
// fp32 -> bf16 pre-convert (each element converted exactly once)
// ---------------------------------------------------------------------------
__global__ void __launch_bounds__(256)
cvt_bf16_kernel(const float* __restrict__ src, __bf16* __restrict__ dst,
                int n8) {
  const int i = blockIdx.x * 256 + threadIdx.x;
  if (i >= n8) return;
  const float4 f0 = ((const float4*)src)[i * 2];
  const float4 f1 = ((const float4*)src)[i * 2 + 1];
  v8bf r;
  r[0]=(__bf16)f0.x; r[1]=(__bf16)f0.y; r[2]=(__bf16)f0.z; r[3]=(__bf16)f0.w;
  r[4]=(__bf16)f1.x; r[5]=(__bf16)f1.y; r[6]=(__bf16)f1.z; r[7]=(__bf16)f1.w;
  ((v8bf*)dst)[i] = r;
}

// ---------------------------------------------------------------------------
// GEMM: out = scale * (A @ W^T + b), A/W bf16, fp32 accumulate.
// Wave: 16(M) x 64(N); block: 8 waves = 128 x 64. W slab (64x32 bf16 = 4KB)
// is shared by all 8 waves -> double-buffered LDS stage via async loads.
// ROPE: fused rotary embedding on head dims 0..31 (tiles 0,1; n0 % 64 == 0).
// STORE_T: store per-head transposed [b][h][d][s] (for V).
// OUTF32: fp32 output (final projection).
// ---------------------------------------------------------------------------
template <bool ROPE, bool STORE_T, bool OUTF32>
__global__ void __launch_bounds__(256)
gemm_kernel(const __bf16* __restrict__ A, const __bf16* __restrict__ W,
            const float* __restrict__ bias, void* __restrict__ outp,
            float scale) {
  __shared__ __bf16 wslab[2][64 * 32];   // 2 x 4KB

  const int tid  = threadIdx.x;
  const int lane = tid & 31;
  const int wave = tid >> 5;
  const int l16  = lane & 15;
  const int half = lane >> 4;
  const int m0   = blockIdx.x * 128 + wave * 16;
  const int n0   = blockIdx.y * 64;

  // cooperative W-slab fetch: thread -> (row, 8-element k segment)
  const int wr   = tid >> 2;          // 0..63
  const int wseg = (tid & 3) * 8;     // 0/8/16/24
  const __bf16* wsrc = W + (size_t)(n0 + wr) * HID + wseg;

  async_copy16(wsrc, &wslab[0][wr * 32 + wseg]);   // k0 = 0 into buf 0

  v8f c[4] = {{}, {}, {}, {}};
  const __bf16* arow = A + (size_t)(m0 + l16) * HID;

  for (int k0 = 0; k0 < HID; k0 += 32) {
    const int  cur  = (k0 >> 5) & 1;
    const bool more = (k0 + 32) < HID;
    if (more) {
      async_copy16(wsrc + k0 + 32, &wslab[cur ^ 1][wr * 32 + wseg]);
      async_wait1();                   // current slab complete (in-order)
    } else {
      async_wait0();
    }
    __syncthreads();

    __builtin_prefetch(arow + k0 + 256, 0, 1);
    // A: row m = l16; elems e <-> k = k0 + 16*half + e
    v16bf a = cat8(*(const v8bf*)(arow + k0 + 16 * half),
                   *(const v8bf*)(arow + k0 + 16 * half + 8));
#pragma unroll
    for (int t = 0; t < 4; ++t) {
      // B: col n = n0+16t+l16; elems e <-> k = k0 + 16*half + e (from LDS)
      v16bf b = *(const v16bf*)(&wslab[cur][(t * 16 + l16) * 32 + 16 * half]);
      c[t] = wmma_bf16(a, b, c[t]);
    }
    __syncthreads();                   // protect slab before next overwrite
  }

#pragma unroll
  for (int t = 0; t < 4; ++t) {
    const float bs = bias[n0 + t * 16 + l16];
#pragma unroll
    for (int j = 0; j < 8; ++j) c[t][j] = (c[t][j] + bs) * scale;
  }

  if (ROPE) {
    // tiles 0/1 are head dims 0..15 / 16..31; same theta index l16
    const float theta = __expf(-LN_BASE_OVER_16 * (float)l16);
#pragma unroll
    for (int j = 0; j < 8; ++j) {
      const int s = (m0 + j + 8 * half) & (SEQ - 1);
      float sn, cs;
      __sincosf((float)s * theta, &sn, &cs);
      const float x0 = c[0][j], x1 = c[1][j];
      c[0][j] = x0 * cs - x1 * sn;
      c[1][j] = x1 * cs + x0 * sn;
    }
  }

  if (OUTF32) {
    float* out = (float*)outp;
#pragma unroll
    for (int j = 0; j < 8; ++j) {
      const size_t row = (size_t)(m0 + j + 8 * half) * HID + n0;
#pragma unroll
      for (int t = 0; t < 4; ++t) out[row + t * 16 + l16] = c[t][j];
    }
  } else if (STORE_T) {
    // [b][h][d][s]: per lane, 8 consecutive s at fixed d -> 16B stores
    __bf16* out = (__bf16*)outp;
    const int    bb    = m0 >> 11;     // / SEQ
    const size_t sbase = (size_t)(m0 & (SEQ - 1)) + 8 * half;
#pragma unroll
    for (int t = 0; t < 4; ++t) {
      const int n = n0 + t * 16 + l16;
      const int hh = n >> 6, dd = n & 63;
      v8bf pk;
#pragma unroll
      for (int j = 0; j < 8; ++j) pk[j] = (__bf16)c[t][j];
      *(v8bf*)(out + (((size_t)bb * HEADS + hh) * KDIM + dd) * SEQ + sbase) = pk;
    }
  } else {
    __bf16* out = (__bf16*)outp;
#pragma unroll
    for (int j = 0; j < 8; ++j) {
      const size_t row = (size_t)(m0 + j + 8 * half) * HID + n0;
#pragma unroll
      for (int t = 0; t < 4; ++t) out[row + t * 16 + l16] = (__bf16)c[t][j];
    }
  }
}

// ---------------------------------------------------------------------------
// Flash attention. One wave = one (b,h) x 16-query tile. q is pre-scaled by
// 1/sqrt(d). Row sums come from a ones-column WMMA (o4); row max via DPP.
// q,k: [b][s][h*64+d] bf16 (roped). vT: [b][h][d][s]. o: [b][s][h*64+d].
// ---------------------------------------------------------------------------
__global__ void __launch_bounds__(256)
attn_kernel(const __bf16* __restrict__ q, const __bf16* __restrict__ k,
            const __bf16* __restrict__ vT, __bf16* __restrict__ o) {
  __shared__ __bf16 plds[8][16 * 32];   // per-wave P bounce (1KB each)

  const int lane = threadIdx.x & 31;
  const int wave = threadIdx.x >> 5;
  const int l16  = lane & 15;
  const int half = lane >> 4;
  const int gw   = blockIdx.x * 8 + wave;
  const int it   = gw & 127;            // SEQ/16 query tiles per (b,h)
  const int bh   = gw >> 7;
  const int b    = bh >> 4;
  const int h    = bh & 15;
  const int i0   = it * 16;

  const __bf16* qrow = q + (size_t)(b * SEQ + i0 + l16) * HID + h * KDIM;
  v16bf qa[2];
#pragma unroll
  for (int c = 0; c < 2; ++c)
    qa[c] = cat8(*(const v8bf*)(qrow + c * 32 + 16 * half),
                 *(const v8bf*)(qrow + c * 32 + 16 * half + 8));

  v16bf ones;
#pragma unroll
  for (int e = 0; e < 16; ++e) ones[e] = (__bf16)1.0f;

  v8f o0 = {}, o1 = {}, o2 = {}, o3 = {}, o4 = {};   // o4 = row sums
  float vm[8];
#pragma unroll
  for (int j = 0; j < 8; ++j) vm[j] = -1e30f;

  const __bf16* kbase = k + (size_t)(b * SEQ) * HID + h * KDIM + 16 * half;
  const __bf16* vbase = vT + ((size_t)(b * HEADS + h) * KDIM) * SEQ + 16 * half;
  __bf16* pl = plds[wave];

  for (int j0 = 0; j0 < SEQ; j0 += 32) {
    __builtin_prefetch(kbase + (size_t)(j0 + 64 + l16) * HID, 0, 1);
    // ---- scores: two 16x16 tiles over key cols j0..j0+31 ----
    v8f s0 = {}, s1 = {};
#pragma unroll
    for (int c = 0; c < 2; ++c) {
      v16bf kb0 = *(const v16bf*)(kbase + (size_t)(j0 + l16) * HID + c * 32);
      v16bf kb1 = *(const v16bf*)(kbase + (size_t)(j0 + 16 + l16) * HID + c * 32);
      s0 = wmma_bf16(qa[c], kb0, s0);
      s1 = wmma_bf16(qa[c], kb1, s1);
    }

    // ---- online softmax (row m = j + 8*half lives in one 16-lane half) ----
    float fac[8];
#pragma unroll
    for (int j = 0; j < 8; ++j) {
      const float mx = rowmax16(fmaxf(s0[j], s1[j]));
      const float nm = fmaxf(vm[j], mx);
      fac[j] = __expf(vm[j] - nm);
      vm[j]  = nm;
      const float p0 = __expf(s0[j] - nm);
      const float p1 = __expf(s1[j] - nm);
      pl[(j + 8 * half) * 32 + l16]      = (__bf16)p0;
      pl[(j + 8 * half) * 32 + 16 + l16] = (__bf16)p1;
    }
#pragma unroll
    for (int j = 0; j < 8; ++j) {
      o0[j] *= fac[j]; o1[j] *= fac[j]; o2[j] *= fac[j]; o3[j] *= fac[j];
      o4[j] *= fac[j];
    }
#if __has_builtin(__builtin_amdgcn_s_wait_dscnt)
    __builtin_amdgcn_s_wait_dscnt(0);   // same-wave DS RAW (backend waits too)
#endif

    // ---- P re-striped C-layout -> A-layout through LDS ----
    const __bf16* pr = pl + l16 * 32;
    v16bf pa = cat8(*(const v8bf*)(pr + 16 * half),
                    *(const v8bf*)(pr + 16 * half + 8));

    // ---- PV + row-sum column ----
    const __bf16* vr = vbase + j0;
    v16bf v0 = *(const v16bf*)(vr + (size_t)(l16) * SEQ);
    v16bf v1 = *(const v16bf*)(vr + (size_t)(16 + l16) * SEQ);
    v16bf v2 = *(const v16bf*)(vr + (size_t)(32 + l16) * SEQ);
    v16bf v3 = *(const v16bf*)(vr + (size_t)(48 + l16) * SEQ);
    o0 = wmma_bf16(pa, v0, o0);
    o1 = wmma_bf16(pa, v1, o1);
    o2 = wmma_bf16(pa, v2, o2);
    o3 = wmma_bf16(pa, v3, o3);
    o4 = wmma_bf16(pa, ones, o4);       // rowsum(P) in every lane
  }

#pragma unroll
  for (int j = 0; j < 8; ++j) {
    const float  inv = 1.0f / o4[j];
    const size_t row = (size_t)(b * SEQ + i0 + j + 8 * half) * HID + h * KDIM;
    o[row + l16]      = (__bf16)(o0[j] * inv);
    o[row + 16 + l16] = (__bf16)(o1[j] * inv);
    o[row + 32 + l16] = (__bf16)(o2[j] * inv);
    o[row + 48 + l16] = (__bf16)(o3[j] * inv);
  }
}

// ---------------------------------------------------------------------------
extern "C" void kernel_launch(void* const* d_in, const int* in_sizes, int n_in,
                              void* d_out, int out_size, void* d_ws,
                              size_t ws_size, hipStream_t stream) {
  (void)in_sizes; (void)n_in; (void)out_size; (void)ws_size;
  const float* query = (const float*)d_in[0];
  const float* value = (const float*)d_in[1];
  const float* Wq    = (const float*)d_in[2];
  const float* bq    = (const float*)d_in[3];
  const float* Wk    = (const float*)d_in[4];
  const float* bk    = (const float*)d_in[5];
  const float* Wv    = (const float*)d_in[6];
  const float* bv    = (const float*)d_in[7];
  const float* Wo    = (const float*)d_in[8];
  const float* bo    = (const float*)d_in[9];
  float* out = (float*)d_out;

  // workspace layout (bytes): 4x act tensors (16MB), 2x bf16 inputs (16MB),
  // 4x bf16 weights (2MB)  => 104MB total
  const size_t TB = (size_t)NTOK * HID * 2;   // 16MB
  const size_t WB = (size_t)HID * HID * 2;    // 2MB
  char* ws = (char*)d_ws;
  __bf16* qbf = (__bf16*)(ws + 0 * TB);
  __bf16* kbf = (__bf16*)(ws + 1 * TB);
  __bf16* vbf = (__bf16*)(ws + 2 * TB);       // per-head transposed
  __bf16* obf = (__bf16*)(ws + 3 * TB);
  __bf16* xq  = (__bf16*)(ws + 4 * TB);
  __bf16* xv  = (__bf16*)(ws + 5 * TB);
  __bf16* wq  = (__bf16*)(ws + 6 * TB + 0 * WB);
  __bf16* wk  = (__bf16*)(ws + 6 * TB + 1 * WB);
  __bf16* wv  = (__bf16*)(ws + 6 * TB + 2 * WB);
  __bf16* wo  = (__bf16*)(ws + 6 * TB + 3 * WB);

  dim3 blk(256);
  const int n8x = NTOK * HID / 8;   // 1,048,576
  const int n8w = HID * HID / 8;    //   131,072
  cvt_bf16_kernel<<<n8x / 256, blk, 0, stream>>>(query, xq, n8x);
  cvt_bf16_kernel<<<n8x / 256, blk, 0, stream>>>(value, xv, n8x);
  cvt_bf16_kernel<<<n8w / 256, blk, 0, stream>>>(Wq, wq, n8w);
  cvt_bf16_kernel<<<n8w / 256, blk, 0, stream>>>(Wk, wk, n8w);
  cvt_bf16_kernel<<<n8w / 256, blk, 0, stream>>>(Wv, wv, n8w);
  cvt_bf16_kernel<<<n8w / 256, blk, 0, stream>>>(Wo, wo, n8w);

  dim3 gproj(NTOK / 128, HID / 64);
  gemm_kernel<true,  false, false><<<gproj, blk, 0, stream>>>(xq, wq, bq, qbf, SCALE);
  gemm_kernel<true,  false, false><<<gproj, blk, 0, stream>>>(xv, wk, bk, kbf, 1.0f);
  gemm_kernel<false, true,  false><<<gproj, blk, 0, stream>>>(xv, wv, bv, vbf, 1.0f);

  attn_kernel<<<dim3(BATCH * HEADS * (SEQ / 16) / 8), blk, 0, stream>>>(
      qbf, kbf, vbf, obf);

  gemm_kernel<false, false, true><<<gproj, blk, 0, stream>>>(obf, wo, bo, out, 1.0f);
}
